// PairsClassifier_73684458930456
// MI455X (gfx1250) — matrix operand order repaired and checked by reading, not compiled, not run on hardware
//
#include <hip/hip_runtime.h>

#define SEQ   128
#define BATCH 2
#define DIN   768
#define PPOS  256
#define DP    1024     // DIN + PPOS
#define DCAT  2048
#define HID   1024
#define NTOT  32768    // B*S*S
#define LDA   40       // padded LDS row stride (elements)
#define LDB   40
#define NK    (DCAT / 32)

typedef __attribute__((ext_vector_type(16))) __bf16 v16bf;
typedef __attribute__((ext_vector_type(8)))  float  v8f;

union Frag16 { uint4 q[2]; v16bf v; };

static __device__ __forceinline__ unsigned short f2bf(float f) {
    unsigned u = __float_as_uint(f);
    u = (u + 0x7FFFu + ((u >> 16) & 1u)) >> 16;   // round-to-nearest-even
    return (unsigned short)u;
}
static __device__ __forceinline__ float sigm(float x) {
    return 1.0f / (1.0f + __expf(-x));
}
static __device__ __forceinline__ float fast_tanh(float x) {
    return 1.0f - 2.0f / (__expf(2.0f * x) + 1.0f);
}

// ---- CDNA5 async global->LDS copy (ASYNCcnt-tracked), inline asm ----
static __device__ __forceinline__ void async_b128(unsigned lds_addr, const void* gptr) {
    asm volatile("global_load_async_to_lds_b128 %0, %1, off"
                 :: "v"(lds_addr), "v"((unsigned long long)gptr) : "memory");
}
static __device__ __forceinline__ void wait_async_le5() {
    asm volatile("s_wait_asynccnt 0x5" ::: "memory");
}
static __device__ __forceinline__ void wait_async_0() {
    asm volatile("s_wait_asynccnt 0x0" ::: "memory");
}

// ---------------- prep kernels ----------------

__global__ __launch_bounds__(256) void build_e_kernel(
    const float* __restrict__ x, const float* __restrict__ pos,
    unsigned short* __restrict__ e)
{
    __shared__ float red[256];
    const int bs = blockIdx.x;                 // 0..255  (b*S + s)
    const float* row = x + (size_t)bs * DIN;
    float p = 0.f;
    for (int k = threadIdx.x; k < DIN; k += 256) { float v = row[k]; p += v * v; }
    red[threadIdx.x] = p;
    __syncthreads();
    for (int s2 = 128; s2 > 0; s2 >>= 1) {
        if (threadIdx.x < s2) red[threadIdx.x] += red[threadIdx.x + s2];
        __syncthreads();
    }
    int idx = (int)sqrtf(red[0]);
    idx = idx < 0 ? 0 : (idx > SEQ - 1 ? SEQ - 1 : idx);
    unsigned short* er = e + (size_t)bs * DP;
    for (int c = threadIdx.x; c < DP; c += 256) {
        float v = (c < DIN) ? row[c] : pos[idx * PPOS + (c - DIN)];
        er[c] = f2bf(v);
    }
}

__global__ __launch_bounds__(256) void cvt_bf16_kernel(
    const float* __restrict__ src, unsigned short* __restrict__ dst, int n)
{
    int i = blockIdx.x * 256 + threadIdx.x;
    int stride = gridDim.x * 256;
    for (; i < n; i += stride) dst[i] = f2bf(src[i]);
}

__global__ __launch_bounds__(256) void bias_combine_kernel(
    const float* __restrict__ a, const float* __restrict__ b,
    float* __restrict__ dst, int n)
{
    int i = blockIdx.x * 256 + threadIdx.x;
    if (i < n) dst[i] = a[i] + b[i];
}

__global__ __launch_bounds__(256) void init_out_kernel(
    float* __restrict__ out, const float* __restrict__ fcb, int n)
{
    int i = blockIdx.x * 256 + threadIdx.x;
    if (i < n) out[i] = fcb[0];
}

// ---------------- fused LSTM-gate GEMM kernels ----------------
// C tile 128 rows x 64 cols (of H), 8 waves (4x2), wave tile 32x32,
// 3 accumulator sets (gates i, c, o).  Double-buffered LDS fed by
// async global->LDS copies; D = sig(o)*tanh(sig(i)*tanh(c)).

__global__ __launch_bounds__(256) void layer0_gemm_kernel(
    const unsigned short* __restrict__ e,
    const unsigned short* __restrict__ wF, const unsigned short* __restrict__ wB,
    const float* __restrict__ biasF, const float* __restrict__ biasB,
    unsigned short* __restrict__ h1)
{
    __shared__ unsigned short As[2][128 * LDA];
    __shared__ unsigned short Bs[2][3][64 * LDB];

    const int tid = threadIdx.x;
    const int dir = blockIdx.z;
    const unsigned short* W = dir ? wB : wF;
    const float* bias = dir ? biasB : biasF;
    const int colBase = blockIdx.x * 64;
    const int rowBase = blockIdx.y * 128;
    const int b  = rowBase >> 14;              // 16384 rows per batch
    const int ii = (rowBase >> 7) & (SEQ - 1); // shared left index i

    const int wave = tid >> 5, lane = tid & 31;
    const int wr = wave >> 1, wc = wave & 1;
    const int l15 = lane & 15, hilane = lane >> 4;

    v8f acc[3][2][2];
#pragma unroll
    for (int g = 0; g < 3; ++g)
#pragma unroll
        for (int mi = 0; mi < 2; ++mi)
#pragma unroll
            for (int ni = 0; ni < 2; ++ni)
#pragma unroll
                for (int q = 0; q < 8; ++q) acc[g][mi][ni][q] = 0.f;

    // staging assignment: A: 2 threads/row x 16 elems; B: 4 threads/row x 8 elems
    const int ar = tid >> 1;
    const int ah = (tid & 1) * 16;        // element offset within K-slab
    const int bc = tid >> 2;
    const int bk = (tid & 3) * 8;

    const unsigned asA[2] = { (unsigned)(size_t)&As[0][ar * LDA + ah],
                              (unsigned)(size_t)&As[1][ar * LDA + ah] };
    unsigned asB[2][3];
#pragma unroll
    for (int bu = 0; bu < 2; ++bu)
#pragma unroll
        for (int g = 0; g < 3; ++g)
            asB[bu][g] = (unsigned)(size_t)&Bs[bu][g][bc * LDB + bk];

    const int gOff[3] = {0, 2 * HID, 3 * HID};

    // issue 5 async b128 copies for buffer bu, K-slab starting at k0
#define ISSUE_TILE0(bu, k0)                                                     \
    {                                                                           \
        const int koff = (k0) + ah;                                             \
        const unsigned short* asrc = (koff < DP)                                \
            ? e + ((size_t)(b * SEQ + ii) * DP + koff)                          \
            : e + ((size_t)(b * SEQ + ar) * DP + (koff - DP));                  \
        async_b128(asA[bu], asrc);                                              \
        async_b128(asA[bu] + 16, asrc + 8);                                     \
        _Pragma("unroll")                                                       \
        for (int g = 0; g < 3; ++g)                                             \
            async_b128(asB[bu][g],                                              \
                W + (size_t)(gOff[g] + colBase + bc) * DCAT + (k0) + bk);       \
    }

    ISSUE_TILE0(0, 0);
    for (int kt = 0; kt < NK; ++kt) {
        const int cur = kt & 1;
        if (kt + 1 < NK) { ISSUE_TILE0(cur ^ 1, (kt + 1) * 32); wait_async_le5(); }
        else             { wait_async_0(); }
        __syncthreads();

        Frag16 afr[2], bfr[3][2];
        {
            const int kbA = hilane ? 8 : 0;
            const int kbB = hilane ? 8 : 0;
#pragma unroll
            for (int mi = 0; mi < 2; ++mi) {
                const int M = wr * 32 + mi * 16 + l15;
                afr[mi].q[0] = *(const uint4*)&As[cur][M * LDA + kbA];
                afr[mi].q[1] = *(const uint4*)&As[cur][M * LDA + kbA + 16];
            }
#pragma unroll
            for (int g = 0; g < 3; ++g)
#pragma unroll
                for (int ni = 0; ni < 2; ++ni) {
                    const int Nn = wc * 32 + ni * 16 + l15;
                    bfr[g][ni].q[0] = *(const uint4*)&Bs[cur][g][Nn * LDB + kbB];
                    bfr[g][ni].q[1] = *(const uint4*)&Bs[cur][g][Nn * LDB + kbB + 16];
                }
        }
#pragma unroll
        for (int g = 0; g < 3; ++g)
#pragma unroll
            for (int ni = 0; ni < 2; ++ni)
#pragma unroll
                for (int mi = 0; mi < 2; ++mi)
                    acc[g][mi][ni] = __builtin_amdgcn_wmma_f32_16x16x32_bf16(
                        false, afr[mi].v, false, bfr[g][ni].v, (short)0,
                        acc[g][mi][ni], false, false);
        __syncthreads();
    }
#undef ISSUE_TILE0

    // ---- epilogue: bias + gating, store h1 (bf16) ----
    float bi[2], bcv[2], bo[2];
#pragma unroll
    for (int ni = 0; ni < 2; ++ni) {
        const int col = colBase + wc * 32 + ni * 16 + l15;
        bi[ni]  = bias[col];
        bcv[ni] = bias[2 * HID + col];
        bo[ni]  = bias[3 * HID + col];
    }
#pragma unroll
    for (int mi = 0; mi < 2; ++mi)
#pragma unroll
        for (int ni = 0; ni < 2; ++ni) {
            const int col = colBase + wc * 32 + ni * 16 + l15;
#pragma unroll
            for (int r = 0; r < 8; ++r) {
                const int row = rowBase + wr * 32 + mi * 16 + r + hilane * 8;
                float gi = acc[0][mi][ni][r] + bi[ni];
                float gc = acc[1][mi][ni][r] + bcv[ni];
                float go = acc[2][mi][ni][r] + bo[ni];
                float cell = sigm(gi) * fast_tanh(gc);
                float hv = sigm(go) * fast_tanh(cell);
                h1[(size_t)row * DCAT + dir * HID + col] = f2bf(hv);
            }
        }
}

__global__ __launch_bounds__(256) void layer1_gemm_kernel(
    const unsigned short* __restrict__ h1,
    const unsigned short* __restrict__ wF, const unsigned short* __restrict__ wB,
    const float* __restrict__ biasF, const float* __restrict__ biasB,
    const float* __restrict__ fcw, float* __restrict__ out)
{
    __shared__ unsigned short As[2][128 * LDA];
    __shared__ unsigned short Bs[2][3][64 * LDB];

    const int tid = threadIdx.x;
    const int dir = blockIdx.z;
    const unsigned short* W = dir ? wB : wF;
    const float* bias = dir ? biasB : biasF;
    const int colBase = blockIdx.x * 64;
    const int rowBase = blockIdx.y * 128;

    const int wave = tid >> 5, lane = tid & 31;
    const int wr = wave >> 1, wc = wave & 1;
    const int l15 = lane & 15, hilane = lane >> 4;

    v8f acc[3][2][2];
#pragma unroll
    for (int g = 0; g < 3; ++g)
#pragma unroll
        for (int mi = 0; mi < 2; ++mi)
#pragma unroll
            for (int ni = 0; ni < 2; ++ni)
#pragma unroll
                for (int q = 0; q < 8; ++q) acc[g][mi][ni][q] = 0.f;

    const int ar = tid >> 1;
    const int ah = (tid & 1) * 16;
    const int bc = tid >> 2;
    const int bk = (tid & 3) * 8;

    const unsigned asA[2] = { (unsigned)(size_t)&As[0][ar * LDA + ah],
                              (unsigned)(size_t)&As[1][ar * LDA + ah] };
    unsigned asB[2][3];
#pragma unroll
    for (int bu = 0; bu < 2; ++bu)
#pragma unroll
        for (int g = 0; g < 3; ++g)
            asB[bu][g] = (unsigned)(size_t)&Bs[bu][g][bc * LDB + bk];

    const int gOff[3] = {0, 2 * HID, 3 * HID};

#define ISSUE_TILE1(bu, k0)                                                     \
    {                                                                           \
        const unsigned short* asrc =                                            \
            h1 + (size_t)(rowBase + ar) * DCAT + (k0) + ah;                     \
        async_b128(asA[bu], asrc);                                              \
        async_b128(asA[bu] + 16, asrc + 8);                                     \
        _Pragma("unroll")                                                       \
        for (int g = 0; g < 3; ++g)                                             \
            async_b128(asB[bu][g],                                              \
                W + (size_t)(gOff[g] + colBase + bc) * DCAT + (k0) + bk);       \
    }

    ISSUE_TILE1(0, 0);
    for (int kt = 0; kt < NK; ++kt) {
        const int cur = kt & 1;
        if (kt + 1 < NK) { ISSUE_TILE1(cur ^ 1, (kt + 1) * 32); wait_async_le5(); }
        else             { wait_async_0(); }
        __syncthreads();

        Frag16 afr[2], bfr[3][2];
        {
            const int kbA = hilane ? 8 : 0;
            const int kbB = hilane ? 8 : 0;
#pragma unroll
            for (int mi = 0; mi < 2; ++mi) {
                const int M = wr * 32 + mi * 16 + l15;
                afr[mi].q[0] = *(const uint4*)&As[cur][M * LDA + kbA];
                afr[mi].q[1] = *(const uint4*)&As[cur][M * LDA + kbA + 16];
            }
#pragma unroll
            for (int g = 0; g < 3; ++g)
#pragma unroll
                for (int ni = 0; ni < 2; ++ni) {
                    const int Nn = wc * 32 + ni * 16 + l15;
                    bfr[g][ni].q[0] = *(const uint4*)&Bs[cur][g][Nn * LDB + kbB];
                    bfr[g][ni].q[1] = *(const uint4*)&Bs[cur][g][Nn * LDB + kbB + 16];
                }
        }
#pragma unroll
        for (int g = 0; g < 3; ++g)
#pragma unroll
            for (int ni = 0; ni < 2; ++ni)
#pragma unroll
                for (int mi = 0; mi < 2; ++mi)
                    acc[g][mi][ni] = __builtin_amdgcn_wmma_f32_16x16x32_bf16(
                        false, afr[mi].v, false, bfr[g][ni].v, (short)0,
                        acc[g][mi][ni], false, false);
        __syncthreads();
    }
#undef ISSUE_TILE1

    // ---- epilogue: gating, fused fc dot, cross-lane reduce, atomicAdd ----
    float bi[2], bcv[2], bo[2], fw[2];
#pragma unroll
    for (int ni = 0; ni < 2; ++ni) {
        const int col = colBase + wc * 32 + ni * 16 + l15;
        bi[ni]  = bias[col];
        bcv[ni] = bias[2 * HID + col];
        bo[ni]  = bias[3 * HID + col];
        fw[ni]  = fcw[dir * HID + col];
    }
#pragma unroll
    for (int mi = 0; mi < 2; ++mi)
#pragma unroll
        for (int r = 0; r < 8; ++r) {
            float val = 0.f;
#pragma unroll
            for (int ni = 0; ni < 2; ++ni) {
                float gi = acc[0][mi][ni][r] + bi[ni];
                float gc = acc[1][mi][ni][r] + bcv[ni];
                float go = acc[2][mi][ni][r] + bo[ni];
                float cell = sigm(gi) * fast_tanh(gc);
                val += sigm(go) * fast_tanh(cell) * fw[ni];
            }
            for (int off = 8; off > 0; off >>= 1)
                val += __shfl_xor(val, off, 16);
            if (l15 == 0) {
                const int row = rowBase + wr * 32 + mi * 16 + r + hilane * 8;
                atomicAdd(&out[row], val);
            }
        }
}

// ---------------- launch ----------------

extern "C" void kernel_launch(void* const* d_in, const int* in_sizes, int n_in,
                              void* d_out, int out_size, void* d_ws, size_t ws_size,
                              hipStream_t stream) {
    const float* x      = (const float*)d_in[0];
    const float* pos    = (const float*)d_in[1];
    const float* w[4]   = {(const float*)d_in[2], (const float*)d_in[5],
                           (const float*)d_in[8], (const float*)d_in[11]};
    const float* bih[4] = {(const float*)d_in[3], (const float*)d_in[6],
                           (const float*)d_in[9], (const float*)d_in[12]};
    const float* bhh[4] = {(const float*)d_in[4], (const float*)d_in[7],
                           (const float*)d_in[10], (const float*)d_in[13]};
    const float* fcw = (const float*)d_in[14];
    const float* fcb = (const float*)d_in[15];
    float* out = (float*)d_out;

    char* ws = (char*)d_ws;
    unsigned short* e    = (unsigned short*)(ws);                        // 512 KB
    float*          bias = (float*)(ws + ((size_t)1 << 20));             // 64 KB
    unsigned short* wbf  = (unsigned short*)(ws + ((size_t)2 << 20));    // 4 x 16 MB
    unsigned short* h1   = (unsigned short*)(ws + ((size_t)66 << 20));   // 128 MB
    const size_t WSLOT = (size_t)4096 * DCAT;                            // 8.4M elems

    build_e_kernel<<<BATCH * SEQ, 256, 0, stream>>>(x, pos, e);
    for (int m = 0; m < 4; ++m) {
        cvt_bf16_kernel<<<4096, 256, 0, stream>>>(w[m], wbf + m * WSLOT, (int)WSLOT);
        bias_combine_kernel<<<16, 256, 0, stream>>>(bih[m], bhh[m], bias + m * 4096, 4096);
    }
    init_out_kernel<<<NTOT / 256, 256, 0, stream>>>(out, fcb, NTOT);

    dim3 grid(HID / 64, NTOT / 128, 2);   // (col tiles, row tiles, dir)
    layer0_gemm_kernel<<<grid, 256, 0, stream>>>(
        e, wbf + 0 * WSLOT, wbf + 1 * WSLOT, bias + 0 * 4096, bias + 1 * 4096, h1);
    layer1_gemm_kernel<<<grid, 256, 0, stream>>>(
        h1, wbf + 2 * WSLOT, wbf + 3 * WSLOT, bias + 2 * 4096, bias + 3 * 4096,
        fcw, out);
}
